// TransformerBackbone_26079041421933
// MI455X (gfx1250) — matrix operand order repaired and checked by reading, not compiled, not run on hardware
//
#include <hip/hip_runtime.h>
#include <hip/hip_bf16.h>
#include <stdint.h>

// ---------------- CDNA5 WMMA types ----------------
typedef __attribute__((ext_vector_type(16))) __bf16 v16bf;
typedef __attribute__((ext_vector_type(8)))  float  v8f;
typedef __attribute__((ext_vector_type(4)))  int    v4i;

union Frag16 { v16bf v; uint4 q[2]; };
union U4     { uint4 u; __bf16 h[8]; };

static __device__ inline __bf16 f2bf(float f)  { return (__bf16)f; }
static __device__ inline float  bf2f(__bf16 h) { return (float)h;  }

// ---- optional gfx1250 async global->LDS path (guarded) ----
#if __has_builtin(__builtin_amdgcn_global_load_async_to_lds_b128) && \
    __has_builtin(__builtin_amdgcn_s_wait_asynccnt)
#define HAVE_ASYNC_LDS 1
#else
#define HAVE_ASYNC_LDS 0
#endif
#define AS1 __attribute__((address_space(1)))
#define AS3 __attribute__((address_space(3)))

// ---------------- problem constants ----------------
#define L_LAYERS 6
#define NH       4
#define WIN      128
#define B_BATCH  64
#define S_SEQ    512
#define E_EMB    256
#define F_FFN    512
#define MMEM     128
#define T_TOT    640   // MMEM + S_SEQ
#define DH       64
#define QKV_F    768   // 3*E

// =====================================================================
// Generic WMMA GEMM:  out[r,n] = sum_k in[r,k] * W[n,k] + bias[n]
// Block tile 128x128, K-step 32, 8 waves (2x4), wave tile 64x32 (4x2 wmma).
// 2-stage software pipeline, double-buffered LDS, 1 barrier / K-step.
// Input rows: r < splitRows -> pA, else pB (concat support).
// =====================================================================
template<bool IN_BF16, bool OUT_BF16, bool GELU>
__global__ __launch_bounds__(256)
void gemm_kernel(const void* __restrict__ pA, const void* __restrict__ pB,
                 int splitRows, const float* __restrict__ Wt,
                 const float* __restrict__ bias, void* __restrict__ outp,
                 int Mrows, int N, int K)
{
    (void)Mrows;
    __shared__ __bf16 As[2][128][40];   // 32 K + 8 pad, 80B row stride (16B aligned)
    __shared__ __bf16 Bs[2][128][40];

    const int tid  = threadIdx.x;
    const int wave = tid >> 5, lane = tid & 31;
    const int half = lane >> 4, l16 = lane & 15;
    const int wy = wave >> 2, wx = wave & 3;
    const int m0 = blockIdx.y * 128;
    const int n0 = blockIdx.x * 128;
    const int lrow = tid >> 1;        // 0..127
    const int lk   = (tid & 1) * 16;  // 0 or 16

    // per-thread source row pointers
    const int gr = m0 + lrow;
    const __bf16* srcAh = nullptr;
    const float*  srcAf = nullptr;
    if (IN_BF16) {
        srcAh = (gr < splitRows) ? ((const __bf16*)pA + (size_t)gr * K)
                                 : ((const __bf16*)pB + (size_t)(gr - splitRows) * K);
    } else {
        srcAf = (gr < splitRows) ? ((const float*)pA + (size_t)gr * K)
                                 : ((const float*)pB + (size_t)(gr - splitRows) * K);
    }
    const float* srcB = Wt + (size_t)(n0 + lrow) * K;

    float4 fa0, fa1, fa2, fa3;        // A staging regs (fp32 path)
    uint4  ha0, ha1;                  // A staging regs (bf16 non-async path)
    float4 fb0, fb1, fb2, fb3;        // B staging regs (weights fp32)

    v8f acc[4][2] = {};
    const int NK = K >> 5;

    auto cvt = [](float4 x, float4 y) -> uint4 {
        U4 u;
        u.h[0]=f2bf(x.x); u.h[1]=f2bf(x.y); u.h[2]=f2bf(x.z); u.h[3]=f2bf(x.w);
        u.h[4]=f2bf(y.x); u.h[5]=f2bf(y.y); u.h[6]=f2bf(y.z); u.h[7]=f2bf(y.w);
        return u.u;
    };
    auto issueA = [&](int it, int buf) {
        const int k = (it << 5) + lk;
        if (IN_BF16) {
#if HAVE_ASYNC_LDS
            // async DMA: global -> LDS, tracked by ASYNCcnt, no VGPR staging
            __builtin_amdgcn_global_load_async_to_lds_b128(
                (AS1 v4i*)(srcAh + k),     (AS3 v4i*)&As[buf][lrow][lk],     0, 0);
            __builtin_amdgcn_global_load_async_to_lds_b128(
                (AS1 v4i*)(srcAh + k + 8), (AS3 v4i*)&As[buf][lrow][lk + 8], 0, 0);
#else
            ha0 = *(const uint4*)(srcAh + k);
            ha1 = *(const uint4*)(srcAh + k + 8);
#endif
        } else {
            fa0 = *(const float4*)(srcAf + k);
            fa1 = *(const float4*)(srcAf + k + 4);
            fa2 = *(const float4*)(srcAf + k + 8);
            fa3 = *(const float4*)(srcAf + k + 12);
        }
    };
    auto issueB = [&](int it) {
        const int k = (it << 5) + lk;
        fb0 = *(const float4*)(srcB + k);
        fb1 = *(const float4*)(srcB + k + 4);
        fb2 = *(const float4*)(srcB + k + 8);
        fb3 = *(const float4*)(srcB + k + 12);
    };
    auto storeTile = [&](int buf) {
        if (IN_BF16) {
#if !HAVE_ASYNC_LDS
            *(uint4*)&As[buf][lrow][lk]     = ha0;
            *(uint4*)&As[buf][lrow][lk + 8] = ha1;
#endif
        } else {
            *(uint4*)&As[buf][lrow][lk]     = cvt(fa0, fa1);
            *(uint4*)&As[buf][lrow][lk + 8] = cvt(fa2, fa3);
        }
        *(uint4*)&Bs[buf][lrow][lk]     = cvt(fb0, fb1);
        *(uint4*)&Bs[buf][lrow][lk + 8] = cvt(fb2, fb3);
#if HAVE_ASYNC_LDS
        if (IN_BF16) __builtin_amdgcn_s_wait_asynccnt(0);
#endif
    };
    auto compute = [&](int buf) {
        Frag16 afr[4], bfr[2];
#pragma unroll
        for (int mt = 0; mt < 4; ++mt) {
            const int r = wy*64 + mt*16 + l16;
            afr[mt].q[0] = *(const uint4*)&As[buf][r][8*half];
            afr[mt].q[1] = *(const uint4*)&As[buf][r][16 + 8*half];
        }
#pragma unroll
        for (int nt = 0; nt < 2; ++nt) {
            const int r = wx*32 + nt*16 + l16;
            bfr[nt].q[0] = *(const uint4*)&Bs[buf][r][8*half];
            bfr[nt].q[1] = *(const uint4*)&Bs[buf][r][16 + 8*half];
        }
#pragma unroll
        for (int mt = 0; mt < 4; ++mt)
#pragma unroll
            for (int nt = 0; nt < 2; ++nt)
                acc[mt][nt] = __builtin_amdgcn_wmma_f32_16x16x32_bf16(
                    false, afr[mt].v, false, bfr[nt].v, (short)0, acc[mt][nt], false, false);
    };

    // ---- pipeline: stage tile 0, then issue(it+1) / compute(it) / store ----
    issueA(0, 0); issueB(0); storeTile(0);
    __syncthreads();
    for (int it = 0; it < NK; ++it) {
        const int cur  = it & 1;
        const bool more = (it + 1) < NK;
        if (more) { issueA(it + 1, cur ^ 1); issueB(it + 1); }
        compute(cur);
        if (more) { storeTile(cur ^ 1); __syncthreads(); }
    }

    // ---- epilogue: bias (+gelu), store fp32 or bf16 ----
#pragma unroll
    for (int mt = 0; mt < 4; ++mt) {
#pragma unroll
        for (int nt = 0; nt < 2; ++nt) {
            const int col = n0 + wx*32 + nt*16 + l16;
            const float bv = bias[col];
#pragma unroll
            for (int i = 0; i < 8; ++i) {
                const int row = m0 + wy*64 + mt*16 + i + 8*half;
                float v = acc[mt][nt][i] + bv;
                if (GELU) {
                    const float x3 = v*v*v;
                    v = 0.5f * v * (1.0f + tanhf(0.7978845608028654f*(v + 0.044715f*x3)));
                }
                const size_t idx = (size_t)row * N + col;
                if (OUT_BF16) ((__bf16*)outp)[idx] = f2bf(v);
                else          ((float*)outp)[idx]  = v;
            }
        }
    }
}

// =====================================================================
// Windowed flash attention, one wave per (b, head, 16-query tile).
// qkv: bf16 (T, B, 768) = [q | k | v]; RoPE applied on the fly.
// Window W=128 and M=W=128 => key range [s0+1, s0+143] covered by
// exactly 5 key tiles of 32 starting at floor(s0/32)*32 (always < T).
// =====================================================================
__global__ __launch_bounds__(128)
void attn_kernel(const __bf16* __restrict__ qkv, const unsigned char* __restrict__ done,
                 float* __restrict__ o)
{
    extern __shared__ char smem[];
    const int wave = threadIdx.x >> 5, lane = threadIdx.x & 31;
    const int half = lane >> 4, l16 = lane & 15;
    const int gid = blockIdx.x * 4 + wave;
    const int s0  = (gid & 31) * 16;
    const int hb  = gid >> 5;
    const int hd  = hb & 3;
    const int b   = hb >> 2;

    __bf16* KQ = (__bf16*)(smem + wave * 11008);            // 32 x 72 bf16 (K, Q staging)
    __bf16* VT = (__bf16*)(smem + wave * 11008 + 4608);     // 64 x 40 bf16 (V transposed)
    __bf16* PB = (__bf16*)(smem + wave * 11008 + 9728);     // 16 x 40 bf16 (P relayout)

    const float il2 = 13.287712379549449f / 64.0f;          // log2(10000)/64
    const float scale = 0.125f;                             // 1/sqrt(64)
    const bool  dn = done[b] != 0;

    // RoPE inverse frequencies, hoisted (unrolled accesses stay in VGPRs)
    float invt[32];
#pragma unroll
    for (int p = 0; p < 32; ++p)
        invt[p] = exp2f(-(float)(2*p) * il2);

    // ---- stage Q (rows s0..s0+15) with RoPE; build A fragments ----
    {
        const int row = lane >> 1;
        const int cbase = (lane & 1) * 4;
        const int t = MMEM + s0 + row;
        const __bf16* src = qkv + (size_t)t * (B_BATCH*QKV_F) + (size_t)b * QKV_F + hd * DH;
        const float pos = (float)t;
#pragma unroll
        for (int cc = 0; cc < 4; ++cc) {
            const int c = cbase + cc;
            U4 in; in.u = *(const uint4*)(src + c*8);
            U4 out;
#pragma unroll
            for (int j = 0; j < 4; ++j) {
                const int p = c*4 + j;
                float sv, cv; __sincosf(pos * invt[p], &sv, &cv);
                const float x1 = bf2f(in.h[2*j]), x2 = bf2f(in.h[2*j+1]);
                out.h[2*j]   = f2bf(x1*cv - x2*sv);
                out.h[2*j+1] = f2bf(x1*sv + x2*cv);
            }
            *(uint4*)&KQ[row*72 + c*8] = out.u;
        }
    }
    Frag16 qf[2];
#pragma unroll
    for (int ds = 0; ds < 2; ++ds) {
        qf[ds].q[0] = *(const uint4*)&KQ[l16*72 + ds*32 + 8*half];
        qf[ds].q[1] = *(const uint4*)&KQ[l16*72 + ds*32 + 16 + 8*half];
    }

    float m_i[8], l_i[8];
    v8f oacc[4] = {};
#pragma unroll
    for (int i = 0; i < 8; ++i) { m_i[i] = -1e30f; l_i[i] = 0.0f; }

    const int t_lo = (s0 >> 5) << 5;
    for (int kt = 0; kt < 5; ++kt) {
        const int t0 = t_lo + kt*32;
        const int t  = t0 + lane;                 // key row staged by this lane

        // prefetch next tile's K/V rows (1 cacheline each: 64 bf16 = 128 B)
        if (kt + 1 < 5) {
            const __bf16* pks = qkv + (size_t)(t + 32) * (B_BATCH*QKV_F)
                                + (size_t)b * QKV_F + E_EMB + hd * DH;
            __builtin_prefetch(pks, 0, 3);
            __builtin_prefetch(pks + E_EMB, 0, 3);
        }

        // ---- stage K (RoPE, row-major) and V (transposed) ----
        {
            const __bf16* ksrc = qkv + (size_t)t * (B_BATCH*QKV_F) + (size_t)b * QKV_F + E_EMB + hd * DH;
            const __bf16* vsrc = ksrc + E_EMB;
            const float pos = (float)t;
#pragma unroll
            for (int c = 0; c < 8; ++c) {
                U4 in; in.u = *(const uint4*)(ksrc + c*8);
                U4 outk;
#pragma unroll
                for (int j = 0; j < 4; ++j) {
                    const int p = c*4 + j;
                    float sv, cv; __sincosf(pos * invt[p], &sv, &cv);
                    const float x1 = bf2f(in.h[2*j]), x2 = bf2f(in.h[2*j+1]);
                    outk.h[2*j]   = f2bf(x1*cv - x2*sv);
                    outk.h[2*j+1] = f2bf(x1*sv + x2*cv);
                }
                *(uint4*)&KQ[lane*72 + c*8] = outk.u;
                U4 vin; vin.u = *(const uint4*)(vsrc + c*8);
#pragma unroll
                for (int j = 0; j < 8; ++j)
                    VT[(c*8 + j)*40 + lane] = vin.h[j];
            }
        }
        // ---- scores S = Q K^T (16 x 32) via 2x2 wmma ----
        v8f sc[2] = {};
#pragma unroll
        for (int nt = 0; nt < 2; ++nt) {
            const int kr = nt*16 + l16;
#pragma unroll
            for (int ds = 0; ds < 2; ++ds) {
                Frag16 bf_;
                bf_.q[0] = *(const uint4*)&KQ[kr*72 + ds*32 + 8*half];
                bf_.q[1] = *(const uint4*)&KQ[kr*72 + ds*32 + 16 + 8*half];
                sc[nt] = __builtin_amdgcn_wmma_f32_16x16x32_bf16(
                    false, qf[ds].v, false, bf_.v, (short)0, sc[nt], false, false);
            }
        }
        // ---- mask + online softmax (row stats across 16 lanes) ----
        float pv[2][8];
#pragma unroll
        for (int i = 0; i < 8; ++i) {
            const int srow = s0 + i + 8*half;     // query index in S
            float v2[2];
            float vmax = -1e30f;
#pragma unroll
            for (int nt = 0; nt < 2; ++nt) {
                const int tg = t0 + nt*16 + l16;
                float v = sc[nt][i] * scale;
                const bool ok = (tg <= MMEM + srow) && (tg > srow) && (tg >= MMEM || !dn);
                v = ok ? v : -1e30f;
                v2[nt] = v;
                vmax = fmaxf(vmax, v);
            }
            for (int mm = 1; mm < 16; mm <<= 1)
                vmax = fmaxf(vmax, __shfl_xor(vmax, mm, 32));
            const float mnew  = fmaxf(m_i[i], vmax);
            const float alpha = __expf(m_i[i] - mnew);
            float rs = 0.0f;
#pragma unroll
            for (int nt = 0; nt < 2; ++nt) {
                const float p = (v2[nt] <= -1e29f) ? 0.0f : __expf(v2[nt] - mnew);
                pv[nt][i] = p;
                rs += p;
            }
            for (int mm = 1; mm < 16; mm <<= 1)
                rs += __shfl_xor(rs, mm, 32);
            l_i[i] = l_i[i]*alpha + rs;
            m_i[i] = mnew;
#pragma unroll
            for (int dt = 0; dt < 4; ++dt)
                oacc[dt][i] *= alpha;
        }
        // ---- P: C-layout -> A-layout via LDS ----
#pragma unroll
        for (int nt = 0; nt < 2; ++nt)
#pragma unroll
            for (int i = 0; i < 8; ++i)
                PB[(i + 8*half)*40 + nt*16 + l16] = f2bf(pv[nt][i]);
        Frag16 pf;
        pf.q[0] = *(const uint4*)&PB[l16*40 + 8*half];
        pf.q[1] = *(const uint4*)&PB[l16*40 + 16 + 8*half];
        // ---- O += P V (16 x 64) via 4 wmma ----
#pragma unroll
        for (int dt = 0; dt < 4; ++dt) {
            Frag16 vf;
            const int vr = dt*16 + l16;
            vf.q[0] = *(const uint4*)&VT[vr*40 + 8*half];
            vf.q[1] = *(const uint4*)&VT[vr*40 + 16 + 8*half];
            oacc[dt] = __builtin_amdgcn_wmma_f32_16x16x32_bf16(
                false, pf.v, false, vf.v, (short)0, oacc[dt], false, false);
        }
    }
    // ---- normalize + store o (S, B, E) fp32 ----
#pragma unroll
    for (int dt = 0; dt < 4; ++dt) {
        const int d = dt*16 + l16;
#pragma unroll
        for (int i = 0; i < 8; ++i) {
            const int srow = s0 + i + 8*half;
            o[(size_t)srow * (B_BATCH*E_EMB) + (size_t)b * E_EMB + hd*DH + d] =
                oacc[dt][i] / l_i[i];
        }
    }
}

// =====================================================================
// h = LayerNorm(h + add) over E=256, one wave per row
// =====================================================================
__global__ __launch_bounds__(256)
void resln_kernel(float* __restrict__ h, const float* __restrict__ add,
                  const float* __restrict__ sc, const float* __restrict__ bi)
{
    const int lane = threadIdx.x & 31;
    const int row  = blockIdx.x * 8 + (threadIdx.x >> 5);
    float* hp = h + (size_t)row * E_EMB;
    const float* ap = add + (size_t)row * E_EMB;
    float v[8];
    float sum = 0.0f;
#pragma unroll
    for (int j = 0; j < 8; ++j) {
        const int e = lane + 32*j;
        v[j] = hp[e] + ap[e];
        sum += v[j];
    }
    for (int mm = 1; mm < 32; mm <<= 1) sum += __shfl_xor(sum, mm, 32);
    const float mean = sum * (1.0f/E_EMB);
    float var = 0.0f;
#pragma unroll
    for (int j = 0; j < 8; ++j) { const float d = v[j] - mean; var += d*d; }
    for (int mm = 1; mm < 32; mm <<= 1) var += __shfl_xor(var, mm, 32);
    const float r = rsqrtf(var * (1.0f/E_EMB) + 1e-5f);
#pragma unroll
    for (int j = 0; j < 8; ++j) {
        const int e = lane + 32*j;
        hp[e] = (v[j] - mean) * r * sc[e] + bi[e];
    }
}

// final LN on the last sequence position only -> out (B, E) fp32
__global__ __launch_bounds__(256)
void final_kernel(const float* __restrict__ h, const float* __restrict__ sc,
                  const float* __restrict__ bi, float* __restrict__ out)
{
    const int lane = threadIdx.x & 31;
    const int b    = blockIdx.x * 8 + (threadIdx.x >> 5);
    const float* hp = h + ((size_t)(S_SEQ-1) * B_BATCH + b) * E_EMB;
    float v[8];
    float sum = 0.0f;
#pragma unroll
    for (int j = 0; j < 8; ++j) { v[j] = hp[lane + 32*j]; sum += v[j]; }
    for (int mm = 1; mm < 32; mm <<= 1) sum += __shfl_xor(sum, mm, 32);
    const float mean = sum * (1.0f/E_EMB);
    float var = 0.0f;
#pragma unroll
    for (int j = 0; j < 8; ++j) { const float d = v[j] - mean; var += d*d; }
    for (int mm = 1; mm < 32; mm <<= 1) var += __shfl_xor(var, mm, 32);
    const float r = rsqrtf(var * (1.0f/E_EMB) + 1e-5f);
#pragma unroll
    for (int j = 0; j < 8; ++j) {
        const int e = lane + 32*j;
        out[(size_t)b * E_EMB + e] = (v[j] - mean) * r * sc[e] + bi[e];
    }
}

// x (B,S,E) -> h (S,B,E)
__global__ void xpose_kernel(const float* __restrict__ x, float* __restrict__ h)
{
    const int s = blockIdx.x, b = blockIdx.y, e = threadIdx.x;
    h[((size_t)s * B_BATCH + b) * E_EMB + e] = x[((size_t)b * S_SEQ + s) * E_EMB + e];
}

// =====================================================================
extern "C" void kernel_launch(void* const* d_in, const int* in_sizes, int n_in,
                              void* d_out, int out_size, void* d_ws, size_t ws_size,
                              hipStream_t stream)
{
    (void)in_sizes; (void)n_in; (void)out_size; (void)ws_size;
    const float* x    = (const float*)d_in[0];
    const float* mem  = (const float*)d_in[1];
    const unsigned char* done = (const unsigned char*)d_in[2];
    const float* Wqkv = (const float*)d_in[3];
    const float* bqkv = (const float*)d_in[4];
    const float* Wo   = (const float*)d_in[5];
    const float* bo   = (const float*)d_in[6];
    const float* W1   = (const float*)d_in[7];
    const float* b1   = (const float*)d_in[8];
    const float* W2   = (const float*)d_in[9];
    const float* b2   = (const float*)d_in[10];
    const float* ln1s = (const float*)d_in[11];
    const float* ln1b = (const float*)d_in[12];
    const float* ln2s = (const float*)d_in[13];
    const float* ln2b = (const float*)d_in[14];
    const float* fns  = (const float*)d_in[15];
    const float* fnb  = (const float*)d_in[16];

    // workspace layout (reused across phases):
    //   h    : fp32  (S,B,E)      33,554,432 B
    //   buf1 : bf16 qkv (T,B,768) 62,914,560 B   (also fp32 GEMM outputs, reused)
    //   buf2 : fp32 o / bf16 f    33,554,432 B
    char* ws = (char*)d_ws;
    float*  h    = (float*)ws;
    __bf16* qkvb = (__bf16*)(ws + 33554432);
    char*   buf2 = ws + 33554432 + 62914560;
    float*  tmpf = (float*)qkvb;   // fp32 reuse of buf1 after attention

    xpose_kernel<<<dim3(S_SEQ, B_BATCH), E_EMB, 0, stream>>>(x, h);

    for (int l = 0; l < L_LAYERS; ++l) {
        const float* Wqkv_l = Wqkv + (size_t)l * QKV_F * E_EMB;
        const float* Wo_l   = Wo   + (size_t)l * E_EMB * E_EMB;
        const float* W1_l   = W1   + (size_t)l * F_FFN * E_EMB;
        const float* W2_l   = W2   + (size_t)l * E_EMB * F_FFN;
        const float* mem_l  = mem  + (size_t)l * MMEM * B_BATCH * E_EMB;

        // qkv = concat(mem[l], h) @ Wqkv^T + bqkv  -> bf16 (T,B,768)
        gemm_kernel<false, true, false>
            <<<dim3(QKV_F/128, (T_TOT*B_BATCH)/128), 256, 0, stream>>>(
            mem_l, h, MMEM*B_BATCH, Wqkv_l, bqkv + l*QKV_F, qkvb,
            T_TOT*B_BATCH, QKV_F, E_EMB);

        // windowed flash attention -> o (S,B,E) fp32 in buf2
        attn_kernel<<<(B_BATCH*NH*(S_SEQ/16))/4, 128, 4*11008, stream>>>(
            qkvb, done, (float*)buf2);

        // o @ Wo^T + bo -> tmpf
        gemm_kernel<false, false, false>
            <<<dim3(E_EMB/128, (S_SEQ*B_BATCH)/128), 256, 0, stream>>>(
            (const float*)buf2, (const float*)buf2, 0, Wo_l, bo + l*E_EMB, tmpf,
            S_SEQ*B_BATCH, E_EMB, E_EMB);

        // h = LN(h + attn_out)
        resln_kernel<<<(S_SEQ*B_BATCH)/8, 256, 0, stream>>>(
            h, tmpf, ln1s + l*E_EMB, ln1b + l*E_EMB);

        // f = gelu(h @ W1^T + b1) -> bf16 in buf2
        gemm_kernel<false, true, true>
            <<<dim3(F_FFN/128, (S_SEQ*B_BATCH)/128), 256, 0, stream>>>(
            h, h, 0, W1_l, b1 + l*F_FFN, buf2, S_SEQ*B_BATCH, F_FFN, E_EMB);

        // f @ W2^T + b2 -> tmpf (bf16 input, async->LDS path if available)
        gemm_kernel<true, false, false>
            <<<dim3(E_EMB/128, (S_SEQ*B_BATCH)/128), 256, 0, stream>>>(
            buf2, buf2, 0, W2_l, b2 + l*E_EMB, tmpf, S_SEQ*B_BATCH, E_EMB, F_FFN);

        // h = LN(h + ffn_out)
        resln_kernel<<<(S_SEQ*B_BATCH)/8, 256, 0, stream>>>(
            h, tmpf, ln2s + l*E_EMB, ln2b + l*E_EMB);
    }

    // final LN on last position -> out (B, E) fp32
    final_kernel<<<B_BATCH/8, 256, 0, stream>>>(h, fns, fnb, (float*)d_out);
}
